// MAEViT2D_20959440404507
// MI455X (gfx1250) — matrix operand order repaired and checked by reading, not compiled, not run on hardware
//
#include <hip/hip_runtime.h>
#include <hip/hip_bf16.h>
#include <math.h>
#include <stdint.h>

// ---------------- model constants ----------------
#define BBATCH 8
#define IMGSZ  128
#define PSZ    4
#define INC    2
#define HPC    32          // IMG/P
#define LTOK   1024        // HPC*HPC
#define LVIS   615
#define LVP    640         // padded visible length (multiple of 64)
#define EDIM   512
#define EHEADS 8
#define DDIM   384
#define DHEADS 6
#define NRELS  3969        // (2*32-1)^2
#define HD     64          // head dim (both enc & dec)

typedef __attribute__((ext_vector_type(16))) __bf16 v16bf;
typedef __attribute__((ext_vector_type(8)))  __bf16 v8bf;
typedef __attribute__((ext_vector_type(8)))  float  v8f;

union V16U { v16bf v; v8bf h[2]; };

__device__ __forceinline__ v8f v8f_zero() {
    v8f z;
    #pragma unroll
    for (int i = 0; i < 8; ++i) z[i] = 0.f;
    return z;
}

__device__ __forceinline__ v8f wmma_bf16(v16bf a, v16bf b, v8f c) {
    return __builtin_amdgcn_wmma_f32_16x16x32_bf16(
        /*neg_a=*/false, a, /*neg_b=*/false, b,
        /*c_mod=*/(short)0, c, /*reuse_a=*/false, /*reuse_b=*/false);
}

// --- CDNA5 async global->LDS copy (16B per lane), tracked by ASYNCcnt ---
__device__ __forceinline__ void async_ld16(void* lds_ptr, const void* gptr) {
    uint32_t lds_off = (uint32_t)(uintptr_t)lds_ptr;     // low 32 bits = LDS offset
    asm volatile("global_load_async_to_lds_b128 %0, %1, off"
                 :: "v"(lds_off), "v"(gptr)
                 : "memory");
}

template <int N>
__device__ __forceinline__ void wait_async() {
#if __has_builtin(__builtin_amdgcn_s_wait_asynccnt)
    __builtin_amdgcn_s_wait_asynccnt((short)N);
#else
    asm volatile("s_wait_asynccnt %0" :: "i"(N) : "memory");
#endif
}

// ---------------- weight convert + transpose ----------------
// W: depth x [K][N] fp32  ->  Wt: depth x [Npad][K] bf16 (rows >= N zero-filled)
__global__ void wcvt_t_kernel(const float* __restrict__ W, __bf16* __restrict__ Wt,
                              int K, int N, int Npad, int total)
{
    int idx = blockIdx.x * 256 + threadIdx.x;
    if (idx >= total) return;
    int k = idx % K;
    int rest = idx / K;
    int n = rest % Npad;
    int d = rest / Npad;
    float v = (n < N) ? W[(size_t)d * K * N + (size_t)k * N + n] : 0.f;
    Wt[(size_t)d * Npad * K + (size_t)n * K + k] = (__bf16)v;
}

// ---------------- generic WMMA GEMM (bf16 in, async double-buffered LDS) ----------------
// C[M,N] = act(A[M,K] @ Bt^T + bias) (+ Res) ; Bt is [Npad][K] bf16 (pre-transposed)
// act: 0=none, 1=exact GELU, 2=clamp(-6,6).  obf: 1 -> write bf16, 0 -> write fp32.
__global__ __launch_bounds__(128) void gemm_wmma_kernel(
    const __bf16* __restrict__ A, const __bf16* __restrict__ Bt,
    const float* __restrict__ bias, const float* __restrict__ Res,
    void* __restrict__ CoutV, int M, int N, int K, int act, int obf)
{
    __shared__ __attribute__((aligned(64))) __bf16 sA[2][64][40];  // [m][k], row 80B
    __shared__ __attribute__((aligned(64))) __bf16 sB[2][64][40];  // [n][k]

    const int t    = threadIdx.x;
    const int wave = t >> 5, lane = t & 31;
    const int lg   = lane >> 4, ln = lane & 15;
    const int m0   = blockIdx.y * 64;
    const int n0   = blockIdx.x * 64;
    const int nK   = K >> 5;

    v8f acc[4];
    #pragma unroll
    for (int j = 0; j < 4; ++j) acc[j] = v8f_zero();

    // stage tile 0
    #pragma unroll
    for (int rep = 0; rep < 2; ++rep) {
        int e = rep * 128 + t;          // 0..255  (64 rows x 4 chunks)
        int r = e >> 2, c8 = (e & 3) * 8;
        async_ld16(&sA[0][r][c8], A  + (size_t)(m0 + r) * K + c8);
        async_ld16(&sB[0][r][c8], Bt + (size_t)(n0 + r) * K + c8);
    }
    wait_async<0>();
    __syncthreads();

    int buf = 0;
    for (int kt = 0; kt < nK; ++kt) {
        if (kt + 1 < nK) {
            int k0 = (kt + 1) * 32;
            #pragma unroll
            for (int rep = 0; rep < 2; ++rep) {
                int e = rep * 128 + t;
                int r = e >> 2, c8 = (e & 3) * 8;
                async_ld16(&sA[buf ^ 1][r][c8], A  + (size_t)(m0 + r) * K + k0 + c8);
                async_ld16(&sB[buf ^ 1][r][c8], Bt + (size_t)(n0 + r) * K + k0 + c8);
            }
        }
        // compute tile kt
        V16U ua;
        ua.h[0] = *(const v8bf*)&sA[buf][wave * 16 + ln][8 * lg];
        ua.h[1] = *(const v8bf*)&sA[buf][wave * 16 + ln][16 + 8 * lg];
        v16bf afrag = ua.v;
        #pragma unroll
        for (int j = 0; j < 4; ++j) {
            V16U ub;
            ub.h[0] = *(const v8bf*)&sB[buf][j * 16 + ln][8 * lg];
            ub.h[1] = *(const v8bf*)&sB[buf][j * 16 + ln][16 + 8 * lg];
            acc[j] = wmma_bf16(afrag, ub.v, acc[j]);
        }
        wait_async<0>();
        __syncthreads();
        buf ^= 1;
    }

    // epilogue: element (m,n): m = m0 + wave*16 + r + 8*lg, n = n0 + j*16 + ln
    #pragma unroll
    for (int j = 0; j < 4; ++j) {
        int n = n0 + j * 16 + ln;
        if (n >= N) continue;
        #pragma unroll
        for (int r = 0; r < 8; ++r) {
            int m = m0 + wave * 16 + r + 8 * lg;
            float v = acc[j][r];
            if (bias) v += bias[n];
            if (act == 1)      v = 0.5f * v * (1.f + erff(v * 0.70710678118f));
            else if (act == 2) v = fminf(fmaxf(v, -6.f), 6.f);
            if (Res) v += Res[(size_t)m * N + n];
            if (obf) ((__bf16*)CoutV)[(size_t)m * N + n] = (__bf16)v;
            else     ((float*)CoutV)[(size_t)m * N + n] = v;
        }
    }
}

// ---------------- fused flash attention with rel-pos bias (bf16 qkv in/out) ----------------
// qkv: [B*Lpad, 3C] bf16, out: [B*Lpad, C] bf16. One block = (b,h) x one 64-query tile.
__global__ __launch_bounds__(128) void attn_wmma_kernel(
    const __bf16* __restrict__ qkv, __bf16* __restrict__ outb,
    const float* __restrict__ rpb, const int* __restrict__ vis,
    int nH, int Lpad, int Lvalid, int C)
{
    __shared__ __attribute__((aligned(64))) __bf16 sQ [64][72]; // [q][d]
    __shared__ __attribute__((aligned(64))) __bf16 sK [64][72]; // [key][d]
    __shared__ __attribute__((aligned(64))) __bf16 sVt[64][72]; // [d][key]
    __shared__ __attribute__((aligned(64))) __bf16 sP [64][72]; // [q][key]

    const int t    = threadIdx.x;
    const int wave = t >> 5, lane = t & 31;
    const int lg   = lane >> 4, ln = lane & 15;
    const int qt   = blockIdx.x;
    const int b    = blockIdx.y / nH, h = blockIdx.y % nH;
    const int C3   = 3 * C;
    const size_t rowbase = (size_t)b * Lpad;

    // stage Q tile via async copy (raw bf16; softmax scale applied to scores later)
    #pragma unroll
    for (int rep = 0; rep < 4; ++rep) {
        int e = rep * 128 + t;           // 0..511 (64 rows x 8 chunks)
        int r = e >> 3, dc = (e & 7) * 8;
        async_ld16(&sQ[r][dc], qkv + (rowbase + qt * 64 + r) * C3 + h * HD + dc);
    }
    wait_async<0>();
    __syncthreads();

    v16bf aq[2];
    {
        int row = wave * 16 + ln;
        #pragma unroll
        for (int ks = 0; ks < 2; ++ks) {
            V16U u;
            u.h[0] = *(const v8bf*)&sQ[row][ks * 32 + 8 * lg];
            u.h[1] = *(const v8bf*)&sQ[row][ks * 32 + 16 + 8 * lg];
            aq[ks] = u.v;
        }
    }

    // per-row query coords (row owned by this lane-group: m = r + 8*lg)
    int rq[8], cq[8];
    #pragma unroll
    for (int r = 0; r < 8; ++r) {
        int qi = qt * 64 + wave * 16 + r + 8 * lg;
        int tq = vis ? ((qi < Lvalid) ? vis[qi] : 0) : qi;
        rq[r] = tq >> 5; cq[r] = tq & 31;
    }

    float mrow[8], lrow[8];
    v8f o[4];
    #pragma unroll
    for (int r = 0; r < 8; ++r) { mrow[r] = -3.0e38f; lrow[r] = 0.f; }
    #pragma unroll
    for (int j = 0; j < 4; ++j) o[j] = v8f_zero();

    const int nKT = Lpad >> 6;
    for (int kt = 0; kt < nKT; ++kt) {
        __syncthreads();   // previous tile fully consumed before restaging
        // K tile via async copy; V tile transposed via vector load + LDS scatter
        #pragma unroll
        for (int rep = 0; rep < 4; ++rep) {
            int e = rep * 128 + t;
            int r = e >> 3, dc = (e & 7) * 8;
            async_ld16(&sK[r][dc], qkv + (rowbase + kt * 64 + r) * C3 + C + h * HD + dc);
        }
        #pragma unroll
        for (int rep = 0; rep < 4; ++rep) {
            int e = rep * 128 + t;
            int kr = e >> 3, dc = (e & 7) * 8;
            v8bf vv = *(const v8bf*)(qkv + (rowbase + kt * 64 + kr) * C3 + 2 * C + h * HD + dc);
            #pragma unroll
            for (int i = 0; i < 8; ++i) sVt[dc + i][kr] = vv[i];
        }
        wait_async<0>();
        __syncthreads();

        // S = (Q K^T) * 0.125 + bias
        v8f s[4];
        #pragma unroll
        for (int j = 0; j < 4; ++j) s[j] = v8f_zero();
        #pragma unroll
        for (int j = 0; j < 4; ++j) {
            #pragma unroll
            for (int ks = 0; ks < 2; ++ks) {
                V16U u;
                u.h[0] = *(const v8bf*)&sK[j * 16 + ln][ks * 32 + 8 * lg];
                u.h[1] = *(const v8bf*)&sK[j * 16 + ln][ks * 32 + 16 + 8 * lg];
                s[j] = wmma_bf16(aq[ks], u.v, s[j]);
            }
        }
        #pragma unroll
        for (int j = 0; j < 4; ++j) {
            int ki = kt * 64 + j * 16 + ln;
            bool valid = ki < Lvalid;
            int tk = vis ? (valid ? vis[ki] : 0) : ki;
            int rk = tk >> 5, ck = tk & 31;
            #pragma unroll
            for (int r = 0; r < 8; ++r) {
                float bv;
                if (valid) {
                    int rel = (rq[r] - rk + 31) * 63 + (cq[r] - ck + 31);
                    bv = rpb[rel * nH + h];
                } else {
                    bv = -1.0e30f;
                }
                s[j][r] = s[j][r] * 0.125f + bv;
            }
        }

        // online softmax (reduce across 16 lanes of each lane-group)
        float alpha[8];
        #pragma unroll
        for (int r = 0; r < 8; ++r) {
            float mx = fmaxf(fmaxf(s[0][r], s[1][r]), fmaxf(s[2][r], s[3][r]));
            for (int off = 1; off < 16; off <<= 1)
                mx = fmaxf(mx, __shfl_xor(mx, off, 16));
            float nm = fmaxf(mrow[r], mx);
            alpha[r] = __expf(mrow[r] - nm);
            mrow[r]  = nm;
            float rs = 0.f;
            #pragma unroll
            for (int j = 0; j < 4; ++j) {
                float p = __expf(s[j][r] - nm);
                s[j][r] = p; rs += p;
            }
            for (int off = 1; off < 16; off <<= 1)
                rs += __shfl_xor(rs, off, 16);
            lrow[r] = lrow[r] * alpha[r] + rs;
        }

        // rescale O and park P (bf16) in LDS to reshape into A-fragments
        #pragma unroll
        for (int j = 0; j < 4; ++j) {
            #pragma unroll
            for (int r = 0; r < 8; ++r) {
                o[j][r] = o[j][r] * alpha[r];
                sP[wave * 16 + r + 8 * lg][j * 16 + ln] = (__bf16)s[j][r];
            }
        }
        __syncthreads();

        // O += P @ V
        v16bf ap[2];
        {
            int row = wave * 16 + ln;
            #pragma unroll
            for (int ks = 0; ks < 2; ++ks) {
                V16U u;
                u.h[0] = *(const v8bf*)&sP[row][ks * 32 + 8 * lg];
                u.h[1] = *(const v8bf*)&sP[row][ks * 32 + 16 + 8 * lg];
                ap[ks] = u.v;
            }
        }
        #pragma unroll
        for (int j = 0; j < 4; ++j) {
            #pragma unroll
            for (int ks = 0; ks < 2; ++ks) {
                V16U u;
                u.h[0] = *(const v8bf*)&sVt[j * 16 + ln][ks * 32 + 8 * lg];
                u.h[1] = *(const v8bf*)&sVt[j * 16 + ln][ks * 32 + 16 + 8 * lg];
                o[j] = wmma_bf16(ap[ks], u.v, o[j]);
            }
        }
    }

    // out = O / l  (bf16)
    #pragma unroll
    for (int j = 0; j < 4; ++j) {
        #pragma unroll
        for (int r = 0; r < 8; ++r) {
            int m  = wave * 16 + r + 8 * lg;
            int qi = qt * 64 + m;
            int d  = j * 16 + ln;
            outb[(rowbase + qi) * C + h * HD + d] = (__bf16)(o[j][r] / lrow[r]);
        }
    }
}

// ---------------- LayerNorm (fp32 in -> bf16 out) ----------------
__global__ __launch_bounds__(256) void ln_kernel(
    const float* __restrict__ xin, const float* __restrict__ g,
    const float* __restrict__ bta, __bf16* __restrict__ yout, int C)
{
    __shared__ float red[256];
    const int row = blockIdx.x;
    const float* xr = xin + (size_t)row * C;
    float s = 0.f;
    for (int c = threadIdx.x; c < C; c += 256) s += xr[c];
    red[threadIdx.x] = s; __syncthreads();
    for (int off = 128; off > 0; off >>= 1) {
        if (threadIdx.x < off) red[threadIdx.x] += red[threadIdx.x + off];
        __syncthreads();
    }
    float mean = red[0] / (float)C;
    __syncthreads();
    float s2 = 0.f;
    for (int c = threadIdx.x; c < C; c += 256) { float d = xr[c] - mean; s2 += d * d; }
    red[threadIdx.x] = s2; __syncthreads();
    for (int off = 128; off > 0; off >>= 1) {
        if (threadIdx.x < off) red[threadIdx.x] += red[threadIdx.x + off];
        __syncthreads();
    }
    float inv = rsqrtf(red[0] / (float)C + 1e-5f);
    __bf16* yr = yout + (size_t)row * C;
    for (int c = threadIdx.x; c < C; c += 256)
        yr[c] = (__bf16)((xr[c] - mean) * inv * g[c] + bta[c]);
}

// ---------------- glue kernels ----------------
__global__ void patchify_gather_kernel(const float* __restrict__ x,
                                       const int* __restrict__ vis,
                                       __bf16* __restrict__ tokA)
{
    int idx = blockIdx.x * 256 + threadIdx.x;
    if (idx >= BBATCH * LVP * 32) return;
    int f = idx & 31, row = idx >> 5;
    int b = row / LVP, lp = row % LVP;
    float v = 0.f;
    if (lp < LVIS) {
        int l = vis[lp];
        int hi = l >> 5, wi = l & 31;
        int c = f & 1, t2 = f >> 1;
        int q = t2 & 3, p = t2 >> 2;
        v = x[(((size_t)b * INC + c) * IMGSZ + (hi * PSZ + p)) * IMGSZ + (wi * PSZ + q)];
    }
    tokA[idx] = (__bf16)v;
}

__global__ void addpos_kernel(float* __restrict__ encX, const float* __restrict__ pe,
                              const int* __restrict__ vis)
{
    int idx = blockIdx.x * 256 + threadIdx.x;
    if (idx >= BBATCH * LVIS * EDIM) return;
    int c = idx & 511, t2 = idx >> 9;
    int lp = t2 % LVIS, b = t2 / LVIS;
    encX[((size_t)(b * LVP + lp)) * EDIM + c] += pe[(size_t)vis[lp] * EDIM + c];
}

__global__ void inv_init_kernel(int* inv)
{
    int i = blockIdx.x * 256 + threadIdx.x;
    if (i < LTOK) inv[i] = -1;
}
__global__ void inv_set_kernel(int* inv, const int* __restrict__ vis)
{
    int i = blockIdx.x * 256 + threadIdx.x;
    if (i < LVIS) inv[vis[i]] = i;
}

__global__ void scatter_dec_kernel(const float* __restrict__ decvis,
                                   const float* __restrict__ mtok,
                                   const float* __restrict__ posd,
                                   const int* __restrict__ inv,
                                   float* __restrict__ decX)
{
    int idx = blockIdx.x * 256 + threadIdx.x;
    if (idx >= BBATCH * LTOK * DDIM) return;
    int c = idx % DDIM, r = idx / DDIM;
    int l = r & 1023, b = r >> 10;
    int vp = inv[l];
    float v = (vp >= 0) ? decvis[((size_t)(b * LVP + vp)) * DDIM + c] : mtok[c];
    decX[idx] = v + posd[(size_t)l * DDIM + c];
}

__global__ void unpatch_kernel(const float* __restrict__ mu, const float* __restrict__ lv,
                               float* __restrict__ outp)
{
    int idx = blockIdx.x * 256 + threadIdx.x;
    if (idx >= BBATCH * IMGSZ * IMGSZ) return;
    int b = idx >> 14, rem = idx & 16383;
    int yy = rem >> 7, xx = rem & 127;
    int hh = yy >> 2, p = yy & 3, w = xx >> 2, q = xx & 3;
    int row = b * LTOK + hh * HPC + w;
    int f = p * PSZ + q;
    outp[idx]                          = mu[(size_t)row * 16 + f];
    outp[BBATCH * IMGSZ * IMGSZ + idx] = lv[(size_t)row * 16 + f];
}

__global__ void mask_kernel(const int* __restrict__ inv, float* __restrict__ outm)
{
    int idx = blockIdx.x * 256 + threadIdx.x;
    if (idx >= BBATCH * LTOK) return;
    outm[idx] = (inv[idx & 1023] >= 0) ? 0.f : 1.f;
}

// ---------------- host orchestration ----------------
static inline int cdiv(int a, int b) { return (a + b - 1) / b; }

extern "C" void kernel_launch(void* const* d_in, const int* in_sizes, int n_in,
                              void* d_out, int out_size, void* d_ws, size_t ws_size,
                              hipStream_t stream)
{
    (void)in_sizes; (void)n_in; (void)out_size; (void)ws_size;

    const float* x          = (const float*)d_in[0];
    const float* enc_w      = (const float*)d_in[1];
    const float* enc_b      = (const float*)d_in[2];
    const float* pos_enc    = (const float*)d_in[3];
    const float* eb_ln1_g   = (const float*)d_in[4];
    const float* eb_ln1_b   = (const float*)d_in[5];
    const float* eb_qkv_w   = (const float*)d_in[6];
    const float* eb_qkv_b   = (const float*)d_in[7];
    const float* eb_rpb     = (const float*)d_in[8];
    const float* eb_proj_w  = (const float*)d_in[9];
    const float* eb_proj_b  = (const float*)d_in[10];
    const float* eb_ln2_g   = (const float*)d_in[11];
    const float* eb_ln2_b   = (const float*)d_in[12];
    const float* eb_fc1_w   = (const float*)d_in[13];
    const float* eb_fc1_b   = (const float*)d_in[14];
    const float* eb_fc2_w   = (const float*)d_in[15];
    const float* eb_fc2_b   = (const float*)d_in[16];
    const float* enc_norm_g = (const float*)d_in[17];
    const float* enc_norm_b = (const float*)d_in[18];
    const float* dec_w      = (const float*)d_in[19];
    const float* dec_b      = (const float*)d_in[20];
    const float* mask_token = (const float*)d_in[21];
    const float* pos_dec    = (const float*)d_in[22];
    const float* db_ln1_g   = (const float*)d_in[23];
    const float* db_ln1_b   = (const float*)d_in[24];
    const float* db_qkv_w   = (const float*)d_in[25];
    const float* db_qkv_b   = (const float*)d_in[26];
    const float* db_rpb     = (const float*)d_in[27];
    const float* db_proj_w  = (const float*)d_in[28];
    const float* db_proj_b  = (const float*)d_in[29];
    const float* db_ln2_g   = (const float*)d_in[30];
    const float* db_ln2_b   = (const float*)d_in[31];
    const float* db_fc1_w   = (const float*)d_in[32];
    const float* db_fc1_b   = (const float*)d_in[33];
    const float* db_fc2_w   = (const float*)d_in[34];
    const float* db_fc2_b   = (const float*)d_in[35];
    const float* dec_norm_g = (const float*)d_in[36];
    const float* dec_norm_b = (const float*)d_in[37];
    const float* mu_w       = (const float*)d_in[38];
    const float* mu_b       = (const float*)d_in[39];
    const float* lv_w       = (const float*)d_in[40];
    const float* lv_b       = (const float*)d_in[41];
    const int*   vis        = (const int*)d_in[42];

    float* out = (float*)d_out;
    float* ws  = (float*)d_ws;

    const int ME = BBATCH * LVP;   // 5120
    const int MD = BBATCH * LTOK;  // 8192

    // ---- workspace layout (all offsets multiples of 4 floats = 16B) ----
    size_t o = 0;
    float* bufX   = ws + o; o += (size_t)MD * DDIM;     // residual stream (fp32)
    float* bufX2  = ws + o; o += (size_t)MD * DDIM;
    float* bufDV  = ws + o; o += (size_t)ME * DDIM;     // decoder-embed of visible tokens
    float* bufMu  = ws + o; o += (size_t)MD * 16;
    float* bufLv  = ws + o; o += (size_t)MD * 16;
    __bf16* tokAh   = (__bf16*)(ws + o); o += (size_t)ME * 32 / 2;
    __bf16* bufLNh  = (__bf16*)(ws + o); o += (size_t)MD * DDIM / 2;
    __bf16* bufBigh = (__bf16*)(ws + o); o += (size_t)MD * 1536 / 2;   // qkv / fc1
    __bf16* bufAtth = (__bf16*)(ws + o); o += (size_t)MD * DDIM / 2;
    // transposed bf16 weights
    __bf16* enc_wT = (__bf16*)(ws + o); o += (size_t)512 * 32 / 2;
    __bf16* eqkvT  = (__bf16*)(ws + o); o += (size_t)6 * 1536 * 512 / 2;
    __bf16* eprojT = (__bf16*)(ws + o); o += (size_t)6 * 512 * 512 / 2;
    __bf16* efc1T  = (__bf16*)(ws + o); o += (size_t)6 * 2048 * 512 / 2;
    __bf16* efc2T  = (__bf16*)(ws + o); o += (size_t)6 * 512 * 2048 / 2;
    __bf16* dec_wT = (__bf16*)(ws + o); o += (size_t)384 * 512 / 2;
    __bf16* dqkvT  = (__bf16*)(ws + o); o += (size_t)4 * 1152 * 384 / 2;
    __bf16* dprojT = (__bf16*)(ws + o); o += (size_t)4 * 384 * 384 / 2;
    __bf16* dfc1T  = (__bf16*)(ws + o); o += (size_t)4 * 1536 * 384 / 2;
    __bf16* dfc2T  = (__bf16*)(ws + o); o += (size_t)4 * 384 * 1536 / 2;
    __bf16* muT    = (__bf16*)(ws + o); o += (size_t)64 * 384 / 2;
    __bf16* lvT    = (__bf16*)(ws + o); o += (size_t)64 * 384 / 2;
    int*    inv    = (int*)(ws + o);

    dim3 blk128(128), blk256(256);

    // ---- one-time (per launch) weight convert+transpose to bf16 [N][K] ----
    #define WCVT(Wsrc, Wdst, Kk, Nn, Np, Dd) \
        wcvt_t_kernel<<<cdiv((Dd)*(Np)*(Kk), 256), blk256, 0, stream>>>( \
            Wsrc, Wdst, Kk, Nn, Np, (Dd)*(Np)*(Kk))
    WCVT(enc_w,    enc_wT, 32,   512,  512,  1);
    WCVT(eb_qkv_w, eqkvT,  512,  1536, 1536, 6);
    WCVT(eb_proj_w,eprojT, 512,  512,  512,  6);
    WCVT(eb_fc1_w, efc1T,  512,  2048, 2048, 6);
    WCVT(eb_fc2_w, efc2T,  2048, 512,  512,  6);
    WCVT(dec_w,    dec_wT, 512,  384,  384,  1);
    WCVT(db_qkv_w, dqkvT,  384,  1152, 1152, 4);
    WCVT(db_proj_w,dprojT, 384,  384,  384,  4);
    WCVT(db_fc1_w, dfc1T,  384,  1536, 1536, 4);
    WCVT(db_fc2_w, dfc2T,  1536, 384,  384,  4);
    WCVT(mu_w,     muT,    384,  16,   64,   1);
    WCVT(lv_w,     lvT,    384,  16,   64,   1);
    #undef WCVT

    // ---- patch embed (visible tokens, padded to 640/batch) ----
    patchify_gather_kernel<<<cdiv(ME * 32, 256), blk256, 0, stream>>>(x, vis, tokAh);
    gemm_wmma_kernel<<<dim3(EDIM / 64, ME / 64), blk128, 0, stream>>>(
        tokAh, enc_wT, enc_b, nullptr, bufX, ME, EDIM, 32, 0, 0);
    addpos_kernel<<<cdiv(BBATCH * LVIS * EDIM, 256), blk256, 0, stream>>>(bufX, pos_enc, vis);

    // ---- encoder blocks ----
    for (int i = 0; i < 6; ++i) {
        ln_kernel<<<ME, blk256, 0, stream>>>(bufX, eb_ln1_g + i * EDIM, eb_ln1_b + i * EDIM, bufLNh, EDIM);
        gemm_wmma_kernel<<<dim3(3 * EDIM / 64, ME / 64), blk128, 0, stream>>>(
            bufLNh, eqkvT + (size_t)i * 1536 * 512, eb_qkv_b + i * 3 * EDIM, nullptr,
            bufBigh, ME, 3 * EDIM, EDIM, 0, 1);
        attn_wmma_kernel<<<dim3(LVP / 64, BBATCH * EHEADS), blk128, 0, stream>>>(
            bufBigh, bufAtth, eb_rpb + (size_t)i * NRELS * EHEADS, vis, EHEADS, LVP, LVIS, EDIM);
        gemm_wmma_kernel<<<dim3(EDIM / 64, ME / 64), blk128, 0, stream>>>(
            bufAtth, eprojT + (size_t)i * 512 * 512, eb_proj_b + i * EDIM, bufX,
            bufX2, ME, EDIM, EDIM, 0, 0);
        ln_kernel<<<ME, blk256, 0, stream>>>(bufX2, eb_ln2_g + i * EDIM, eb_ln2_b + i * EDIM, bufLNh, EDIM);
        gemm_wmma_kernel<<<dim3(4 * EDIM / 64, ME / 64), blk128, 0, stream>>>(
            bufLNh, efc1T + (size_t)i * 2048 * 512, eb_fc1_b + i * 4 * EDIM, nullptr,
            bufBigh, ME, 4 * EDIM, EDIM, 1, 1);
        gemm_wmma_kernel<<<dim3(EDIM / 64, ME / 64), blk128, 0, stream>>>(
            bufBigh, efc2T + (size_t)i * 512 * 2048, eb_fc2_b + i * EDIM, bufX2,
            bufX, ME, EDIM, 4 * EDIM, 0, 0);
    }

    // ---- encoder norm + decoder embed + scatter ----
    ln_kernel<<<ME, blk256, 0, stream>>>(bufX, enc_norm_g, enc_norm_b, bufLNh, EDIM);
    gemm_wmma_kernel<<<dim3(DDIM / 64, ME / 64), blk128, 0, stream>>>(
        bufLNh, dec_wT, dec_b, nullptr, bufDV, ME, DDIM, EDIM, 0, 0);
    inv_init_kernel<<<cdiv(LTOK, 256), blk256, 0, stream>>>(inv);
    inv_set_kernel<<<cdiv(LVIS, 256), blk256, 0, stream>>>(inv, vis);
    scatter_dec_kernel<<<cdiv(MD * DDIM, 256), blk256, 0, stream>>>(
        bufDV, mask_token, pos_dec, inv, bufX);

    // ---- decoder blocks ----
    for (int i = 0; i < 4; ++i) {
        ln_kernel<<<MD, blk256, 0, stream>>>(bufX, db_ln1_g + i * DDIM, db_ln1_b + i * DDIM, bufLNh, DDIM);
        gemm_wmma_kernel<<<dim3(3 * DDIM / 64, MD / 64), blk128, 0, stream>>>(
            bufLNh, dqkvT + (size_t)i * 1152 * 384, db_qkv_b + i * 3 * DDIM, nullptr,
            bufBigh, MD, 3 * DDIM, DDIM, 0, 1);
        attn_wmma_kernel<<<dim3(LTOK / 64, BBATCH * DHEADS), blk128, 0, stream>>>(
            bufBigh, bufAtth, db_rpb + (size_t)i * NRELS * DHEADS, nullptr, DHEADS, LTOK, LTOK, DDIM);
        gemm_wmma_kernel<<<dim3(DDIM / 64, MD / 64), blk128, 0, stream>>>(
            bufAtth, dprojT + (size_t)i * 384 * 384, db_proj_b + i * DDIM, bufX,
            bufX2, MD, DDIM, DDIM, 0, 0);
        ln_kernel<<<MD, blk256, 0, stream>>>(bufX2, db_ln2_g + i * DDIM, db_ln2_b + i * DDIM, bufLNh, DDIM);
        gemm_wmma_kernel<<<dim3(4 * DDIM / 64, MD / 64), blk128, 0, stream>>>(
            bufLNh, dfc1T + (size_t)i * 1536 * 384, db_fc1_b + i * 4 * DDIM, nullptr,
            bufBigh, MD, 4 * DDIM, DDIM, 1, 1);
        gemm_wmma_kernel<<<dim3(DDIM / 64, MD / 64), blk128, 0, stream>>>(
            bufBigh, dfc2T + (size_t)i * 384 * 1536, db_fc2_b + i * DDIM, bufX2,
            bufX, MD, DDIM, 4 * DDIM, 0, 0);
    }

    // ---- heads + unpatchify + mask ----
    ln_kernel<<<MD, blk256, 0, stream>>>(bufX, dec_norm_g, dec_norm_b, bufLNh, DDIM);
    gemm_wmma_kernel<<<dim3(1, MD / 64), blk128, 0, stream>>>(
        bufLNh, muT, mu_b, nullptr, bufMu, MD, 16, DDIM, 0, 0);
    gemm_wmma_kernel<<<dim3(1, MD / 64), blk128, 0, stream>>>(
        bufLNh, lvT, lv_b, nullptr, bufLv, MD, 16, DDIM, 2, 0);
    unpatch_kernel<<<cdiv(BBATCH * IMGSZ * IMGSZ, 256), blk256, 0, stream>>>(bufMu, bufLv, out);
    mask_kernel<<<cdiv(BBATCH * LTOK, 256), blk256, 0, stream>>>(
        inv, out + 2 * BBATCH * IMGSZ * IMGSZ);
}
